// InnerModel_86509231276050
// MI455X (gfx1250) — matrix operand order repaired and checked by reading, not compiled, not run on hardware
//
#include <hip/hip_runtime.h>
#include <hip/hip_bf16.h>
#include <math.h>

typedef _Float16 half_t;
typedef __attribute__((ext_vector_type(16))) _Float16 v16h;
typedef __attribute__((ext_vector_type(8)))  _Float16 v8h;
typedef __attribute__((ext_vector_type(8)))  float    v8f;
typedef int vsi4 __attribute__((vector_size(16)));   // builtin-compatible 4 x i32 vector

union Frag16 { v16h v; v8h h8[2]; half_t e[16]; };

#if __has_builtin(__builtin_amdgcn_global_load_async_to_lds_b128) && __has_builtin(__builtin_amdgcn_s_wait_asynccnt)
#define HAVE_ASYNC_LDS 1
#endif

static __device__ __forceinline__ float elu_f(float x)     { return x > 0.f ? x : (__expf(x) - 1.f); }
static __device__ __forceinline__ float relu_f(float x)    { return x > 0.f ? x : 0.f; }
static __device__ __forceinline__ float sigmoid_f(float x) { return 1.f / (1.f + __expf(-x)); }

// ---------------- LayerNorm over 576 features ----------------
__global__ void ln_kernel(const float* __restrict__ x, const float* __restrict__ g,
                          const float* __restrict__ be, float* __restrict__ out) {
  __shared__ float s1[256], s2[256];
  int b = blockIdx.x, t = threadIdx.x;
  const float* row = x + b * 576;
  float a = 0.f, q = 0.f;
  for (int i = t; i < 576; i += 256) { float v = row[i]; a += v; q += v * v; }
  s1[t] = a; s2[t] = q; __syncthreads();
  for (int o = 128; o > 0; o >>= 1) {
    if (t < o) { s1[t] += s1[t + o]; s2[t] += s2[t + o]; }
    __syncthreads();
  }
  float mean = s1[0] * (1.f / 576.f);
  float var  = s2[0] * (1.f / 576.f) - mean * mean;
  float rstd = rsqrtf(var + 1e-3f);
  for (int i = t; i < 576; i += 256)
    out[b * 576 + i] = (row[i] - mean) * rstd * g[i] + be[i];
}

// ---------------- conv1 only: 4x4 s2 VALID, scalar, f16 out (dims compile-time) ----------------
template<int H, int W, int Cin, int OH, int OW, int Cout>
__global__ void conv_valid_s2_elu_h(const float* __restrict__ in, const float* __restrict__ w,
                                    const float* __restrict__ bias, half_t* __restrict__ out,
                                    int total) {
  int i = blockIdx.x * blockDim.x + threadIdx.x;
  if (i >= total) return;
  unsigned u = (unsigned)i;
  int co = u % Cout; unsigned t = u / Cout;
  int ox = t % OW; t /= OW;
  int oy = t % OH; int b = (int)(t / OH);
  float acc = bias[co];
#pragma unroll
  for (int ky = 0; ky < 4; ++ky)
#pragma unroll
    for (int kx = 0; kx < 4; ++kx) {
      const float* ip = in + ((size_t)(b * H + oy * 2 + ky) * W + ox * 2 + kx) * Cin;
      const float* wp = w + ((size_t)(ky * 4 + kx) * Cin) * Cout + co;
      for (int ci = 0; ci < Cin; ++ci) acc += ip[ci] * wp[(size_t)ci * Cout];
    }
  out[i] = (half_t)elu_f(acc);
}

// ---------------- c3 only: 4x4 s1 SAME, Cout=1, scalar, f32 (dims compile-time) ----------------
template<int H, int W, int Cin, int Cout>
__global__ void conv_same_s1_f(const float* __restrict__ in, const float* __restrict__ w,
                               const float* __restrict__ bias, float* __restrict__ out,
                               int total) {
  int i = blockIdx.x * blockDim.x + threadIdx.x;
  if (i >= total) return;
  unsigned u = (unsigned)i;
  int co = u % Cout; unsigned t = u / Cout;
  int ox = t % W; t /= W;
  int oy = t % H; int b = (int)(t / H);
  float acc = bias[co];
#pragma unroll
  for (int ky = 0; ky < 4; ++ky) {
    int iy = oy + ky - 1;
    if (iy < 0 || iy >= H) continue;
#pragma unroll
    for (int kx = 0; kx < 4; ++kx) {
      int ix = ox + kx - 1;
      if (ix < 0 || ix >= W) continue;
      const float* ip = in + ((size_t)(b * H + iy) * W + ix) * Cin;
      const float* wp = w + ((size_t)(ky * 4 + kx) * Cin) * Cout + co;
      for (int ci = 0; ci < Cin; ++ci) acc += ip[ci] * wp[(size_t)ci * Cout];
    }
  }
  out[i] = acc;
}

// ---------------- 2x bilinear upsample (dims compile-time) ----------------
template<typename InT, typename OutT, int Hin, int C>
__global__ void bilinear2x(const InT* __restrict__ in, OutT* __restrict__ out, int total) {
  int i = blockIdx.x * blockDim.x + threadIdx.x;
  if (i >= total) return;
  const int Hout = Hin * 2;
  unsigned u = (unsigned)i;
  int c = u % C; unsigned t = u / C;
  int ox = t % Hout; t /= Hout;
  int oy = t % Hout; int b = (int)(t / Hout);
  float sy = (oy + 0.5f) * 0.5f - 0.5f;
  float sx = (ox + 0.5f) * 0.5f - 0.5f;
  int y0 = (int)floorf(sy); float fy = sy - (float)y0;
  int x0 = (int)floorf(sx); float fx = sx - (float)x0;
  int y0c = min(max(y0, 0), Hin - 1), y1c = min(max(y0 + 1, 0), Hin - 1);
  int x0c = min(max(x0, 0), Hin - 1), x1c = min(max(x0 + 1, 0), Hin - 1);
  const InT* base = in + (size_t)b * Hin * Hin * C;
  float v00 = (float)base[((size_t)y0c * Hin + x0c) * C + c];
  float v01 = (float)base[((size_t)y0c * Hin + x1c) * C + c];
  float v10 = (float)base[((size_t)y1c * Hin + x0c) * C + c];
  float v11 = (float)base[((size_t)y1c * Hin + x1c) * C + c];
  out[i] = (OutT)((1.f - fy) * ((1.f - fx) * v00 + fx * v01) +
                  fy * ((1.f - fx) * v10 + fx * v11));
}

// ---------------- f32 -> f16 conversions ----------------
__global__ void cvt_f16(const float* __restrict__ s, half_t* __restrict__ d, int n) {
  int i = blockIdx.x * blockDim.x + threadIdx.x;
  if (i < n) d[i] = (half_t)s[i];
}
__global__ void pad_f16(const float* __restrict__ s, half_t* __restrict__ d,
                        int rows, int cols, int colsP, int total) {
  int i = blockIdx.x * blockDim.x + threadIdx.x;
  if (i >= total) return;
  int c = i % colsP, r = i / colsP;
  d[i] = (r < rows && c < cols) ? (half_t)s[(size_t)r * cols + c] : (half_t)0.f;
}

// ---------------- WMMA GEMM: C = act(A(MxKp) * B(KpxN) + bias), wave per 16x16 tile ----------------
template<int ACT, int N, int Kp, typename OutT> // ACT: 0=none 1=relu 2=elu
__global__ __launch_bounds__(32) void gemm16(const half_t* __restrict__ A, const half_t* __restrict__ Bw,
                                             const float* __restrict__ bias, OutT* __restrict__ C) {
  int lane = threadIdx.x & 31;
  int hi = lane >> 4, nl = lane & 15;
  int m0 = blockIdx.x * 16;
  int n  = blockIdx.y * 16 + nl;
  const half_t* arow = A + (size_t)(m0 + nl) * Kp;
  v8f acc = {};
  for (int k0 = 0; k0 < Kp; k0 += 32) {
    Frag16 a, bm;
    a.h8[0] = *(const v8h*)(arow + k0 + hi * 16);
    a.h8[1] = *(const v8h*)(arow + k0 + hi * 16 + 8);
    const half_t* bp = Bw + (size_t)(k0 + hi * 16) * N + n;
#pragma unroll
    for (int j = 0; j < 16; ++j) bm.e[j] = bp[(size_t)j * N];
    acc = __builtin_amdgcn_wmma_f32_16x16x32_f16(false, a.v, false, bm.v, (short)0, acc, false, false);
  }
  float bb = bias[n];
#pragma unroll
  for (int r = 0; r < 8; ++r) {
    int m = m0 + r + hi * 8;
    float v = acc[r] + bb;
    if (ACT == 1) v = relu_f(v);
    else if (ACT == 2) v = elu_f(v);
    C[(size_t)m * N + n] = (OutT)v;
  }
}

// ---------------- Implicit-GEMM 4x4 stride-2 VALID conv + ELU (f16 in/out) ----------------
// Weights flattened HWIO => row-major (16*Cin) x Cout. Cin power of two >= 8.
template<int H, int W, int Cin, int OH, int OW, int Cout>
__global__ __launch_bounds__(32) void convg_s2_elu(const half_t* __restrict__ inF,
                                                   const half_t* __restrict__ wF,
                                                   const float* __restrict__ bias,
                                                   half_t* __restrict__ outF) {
  const int Kp = 16 * Cin;
  int lane = threadIdx.x & 31;
  int hi = lane >> 4, nl = lane & 15;
  int m0 = blockIdx.x * 16;
  int n  = blockIdx.y * 16 + nl;
  unsigned mA = (unsigned)(m0 + nl);
  int oxA = mA % OW; unsigned tA = mA / OW;
  int oyA = tA % OH; int bA = (int)(tA / OH);
  const half_t* ibase = inF + ((size_t)(bA * H + oyA * 2) * W + oxA * 2) * Cin;
  v8f acc = {};
  for (int k0 = 0; k0 < Kp; k0 += 32) {
    Frag16 a, bm;
    int kb0 = k0 + hi * 16;
    int kb1 = kb0 + 8;
    int p0 = kb0 / Cin, c0 = kb0 % Cin;
    int p1 = kb1 / Cin, c1 = kb1 % Cin;
    a.h8[0] = *(const v8h*)(ibase + ((size_t)(p0 >> 2) * W + (p0 & 3)) * Cin + c0);
    a.h8[1] = *(const v8h*)(ibase + ((size_t)(p1 >> 2) * W + (p1 & 3)) * Cin + c1);
    const half_t* bp = wF + (size_t)(k0 + hi * 16) * Cout + n;
#pragma unroll
    for (int j = 0; j < 16; ++j) bm.e[j] = bp[(size_t)j * Cout];
    acc = __builtin_amdgcn_wmma_f32_16x16x32_f16(false, a.v, false, bm.v, (short)0, acc, false, false);
  }
  float bb = bias[n];
#pragma unroll
  for (int r = 0; r < 8; ++r) {
    unsigned m = (unsigned)(m0 + r + hi * 8);
    int ox = m % OW; unsigned t = m / OW;
    int oy = t % OH; int b = (int)(t / OH);
    outF[((size_t)(b * OH + oy) * OW + ox) * Cout + n] = (half_t)elu_f(acc[r] + bb);
  }
}

// ---------------- Implicit-GEMM 4x4 stride-1 SAME conv + ReLU (f16 in/out) ----------------
template<int H, int W, int Cin, int Cout>
__global__ __launch_bounds__(32) void convg_same_relu(const half_t* __restrict__ inF,
                                                      const half_t* __restrict__ wF,
                                                      const float* __restrict__ bias,
                                                      half_t* __restrict__ outF) {
  const int Kp = 16 * Cin;
  int lane = threadIdx.x & 31;
  int hi = lane >> 4, nl = lane & 15;
  int m0 = blockIdx.x * 16;
  int n  = blockIdx.y * 16 + nl;
  unsigned mA = (unsigned)(m0 + nl);
  int oxA = mA % W; unsigned tA = mA / W;
  int oyA = tA % H; int bA = (int)(tA / H);
  const half_t* bbase = inF + (size_t)bA * H * W * Cin;
  v8f acc = {};
  const v8h vz = {};
  for (int k0 = 0; k0 < Kp; k0 += 32) {
    Frag16 a, bm;
    int kb0 = k0 + hi * 16;
    int kb1 = kb0 + 8;
    int p0 = kb0 / Cin, c0 = kb0 % Cin;
    int p1 = kb1 / Cin, c1 = kb1 % Cin;
    int iy0 = oyA + (p0 >> 2) - 1, ix0 = oxA + (p0 & 3) - 1;
    int iy1 = oyA + (p1 >> 2) - 1, ix1 = oxA + (p1 & 3) - 1;
    bool ok0 = (iy0 >= 0) & (iy0 < H) & (ix0 >= 0) & (ix0 < W);
    bool ok1 = (iy1 >= 0) & (iy1 < H) & (ix1 >= 0) & (ix1 < W);
    a.h8[0] = ok0 ? *(const v8h*)(bbase + ((size_t)iy0 * W + ix0) * Cin + c0) : vz;
    a.h8[1] = ok1 ? *(const v8h*)(bbase + ((size_t)iy1 * W + ix1) * Cin + c1) : vz;
    const half_t* bp = wF + (size_t)(k0 + hi * 16) * Cout + n;
#pragma unroll
    for (int j = 0; j < 16; ++j) bm.e[j] = bp[(size_t)j * Cout];
    acc = __builtin_amdgcn_wmma_f32_16x16x32_f16(false, a.v, false, bm.v, (short)0, acc, false, false);
  }
  float bb = bias[n];
#pragma unroll
  for (int r = 0; r < 8; ++r) {
    unsigned m = (unsigned)(m0 + r + hi * 8);
    int ox = m % W; unsigned t = m / W;
    int oy = t % H; int b = (int)(t / H);
    outF[((size_t)(b * H + oy) * W + ox) * Cout + n] = (half_t)relu_f(acc[r] + bb);
  }
}

// ---------------- Persistent RNN: 512 timesteps, h in LDS, weights in WMMA B-fragments ----------------
__global__ __launch_bounds__(512) void rnn_kernel(const float* __restrict__ x512,   // (512,512)
                                                  const half_t* __restrict__ wrh,   // (256,256) padded f16
                                                  const float* __restrict__ wrin,   // (250)
                                                  const float* __restrict__ brv,    // (250)
                                                  const float* __restrict__ h0,     // (512,250)
                                                  half_t* __restrict__ h16,         // (512,256) f16 out
                                                  float* __restrict__ theta) {      // (512,250) out
  __shared__ __align__(16) half_t hL[16 * 256];   // 8 KB hidden-state tile
  __shared__ __align__(16) float xL[16 * 512];    // 32 KB input tile
  int tid = threadIdx.x;
  int lane = tid & 31, wave = tid >> 5;           // 16 waves, wave = N-tile
  int hi = lane >> 4, nl = lane & 15;
  int n = wave * 16 + nl;
  int b0 = blockIdx.x * 16;

#ifdef HAVE_ASYNC_LDS
  {
    // x-tile rows are contiguous: async-DMA 32 KB straight into LDS
    const float* src = x512 + (size_t)b0 * 512;
    for (int i = tid * 4; i < 16 * 512; i += 512 * 4) {
      __builtin_amdgcn_global_load_async_to_lds_b128(
          (__attribute__((address_space(1))) vsi4*)((float*)src + i),
          (__attribute__((address_space(3))) vsi4*)(&xL[i]), 0, 0);
    }
  }
#else
  for (int i = tid; i < 16 * 512; i += 512) {
    int r = i >> 9, t = i & 511;
    xL[i] = x512[(size_t)(b0 + r) * 512 + t];
  }
#endif
  for (int i = tid; i < 16 * 256; i += 512) {
    int r = i >> 8, c = i & 255;
    hL[i] = (c < 250) ? (half_t)h0[(size_t)(b0 + r) * 250 + c] : (half_t)0.f;
  }
  float win = (n < 250) ? wrin[n] : 0.f;
  float bb  = (n < 250) ? brv[n]  : 0.f;

  // Recurrent weights: 8 K-fragments for this wave's 16 columns, resident in VGPRs
  Frag16 bf[8];
#pragma unroll
  for (int kt = 0; kt < 8; ++kt) {
    const half_t* bp = wrh + (size_t)(kt * 32 + hi * 16) * 256 + n;
#pragma unroll
    for (int j = 0; j < 16; ++j) bf[kt].e[j] = bp[(size_t)j * 256];
  }
#ifdef HAVE_ASYNC_LDS
  __builtin_amdgcn_s_wait_asynccnt(0);
#endif
  __syncthreads();

  const half_t* arow = hL + nl * 256;
  for (int t = 0; t < 512; ++t) {
    v8f acc = {};
#pragma unroll
    for (int kt = 0; kt < 8; ++kt) {
      Frag16 a;
      a.h8[0] = *(const v8h*)(arow + kt * 32 + hi * 16);
      a.h8[1] = *(const v8h*)(arow + kt * 32 + hi * 16 + 8);
      acc = __builtin_amdgcn_wmma_f32_16x16x32_f16(false, a.v, false, bf[kt].v, (short)0, acc, false, false);
    }
    __syncthreads();  // all reads of h done before overwrite
#pragma unroll
    for (int r = 0; r < 8; ++r) {
      int m = r + hi * 8;
      float p = acc[r] + xL[m * 512 + t] * win + bb;
      hL[m * 256 + n] = (half_t)sigmoid_f(p);
    }
    __syncthreads();  // h_new visible before next step's reads
  }

  for (int i = tid; i < 16 * 256; i += 512) {
    int r = i >> 8, c = i & 255;
    half_t v = hL[i];
    h16[(size_t)(b0 + r) * 256 + c] = v;
    if (c < 250) theta[(size_t)(b0 + r) * 250 + c] = (float)v;
  }
}

extern "C" void kernel_launch(void* const* d_in, const int* in_sizes, int n_in,
                              void* d_out, int out_size, void* d_ws, size_t ws_size,
                              hipStream_t stream) {
  const float* inp  = (const float*)d_in[0];
  const float* th0  = (const float*)d_in[1];
  const float* lng  = (const float*)d_in[2];
  const float* lnb  = (const float*)d_in[3];
  const float* w1   = (const float*)d_in[4];  const float* b1   = (const float*)d_in[5];
  const float* w2   = (const float*)d_in[6];  const float* b2   = (const float*)d_in[7];
  const float* w3   = (const float*)d_in[8];  const float* b3   = (const float*)d_in[9];
  const float* we   = (const float*)d_in[10]; const float* be   = (const float*)d_in[11];
  const float* wrin = (const float*)d_in[12]; const float* wrh  = (const float*)d_in[13];
  const float* brv  = (const float*)d_in[14];
  const float* wd1  = (const float*)d_in[15]; const float* bd1  = (const float*)d_in[16];
  const float* wd2  = (const float*)d_in[17]; const float* bd2  = (const float*)d_in[18];
  const float* wdc1 = (const float*)d_in[19]; const float* bdc1 = (const float*)d_in[20];
  const float* wdc2 = (const float*)d_in[21]; const float* bdc2 = (const float*)d_in[22];
  const float* wdc3 = (const float*)d_in[23]; const float* bdc3 = (const float*)d_in[24];

  char* ws = (char*)d_ws;
  half_t* A1H  = (half_t*)(ws + 0);          // 512x11x11x32 f16
  half_t* W2H  = (half_t*)(ws + 3964928);    // 512x64 f16
  half_t* A2H  = (half_t*)(ws + 4030464);    // 512x4x4x64 f16
  half_t* W3H  = (half_t*)(ws + 5079040);    // 1024x128 f16
  half_t* A3H  = (half_t*)(ws + 5341184);    // 512x128 f16
  half_t* WEH  = (half_t*)(ws + 5472256);    // 128x512 f16
  float*  X512 = (float*)(ws + 5603328);     // 512x512 f32
  half_t* WRHH = (half_t*)(ws + 6651904);    // 256x256 f16 (wr_h padded)
  half_t* H16  = (half_t*)(ws + 6782976);    // 512x256 f16
  half_t* WD1H = (half_t*)(ws + 7045120);    // 256x512 f16 (wd1 padded)
  half_t* D1H  = (half_t*)(ws + 7307264);    // 512x512 f16
  half_t* WD2H = (half_t*)(ws + 7831552);    // 512x1152 f16
  float*  D2   = (float*)(ws + 9011200);     // 512x1152 f32 (== y3 512x3x3x128)
  float*  XN   = (float*)(ws + 11370496);    // 512x576 f32
  half_t* WC1H = (half_t*)(ws + 12550144);   // 2048x64 f16
  half_t* WC2H = (half_t*)(ws + 12812288);   // 1024x32 f16
  half_t* R6H  = (half_t*)(ws + 12877824);   // 512x6x6x128 f16
  half_t* C1H  = (half_t*)(ws + 17596416);   // 512x6x6x64 f16
  half_t* R12H = (half_t*)(ws + 19955712);   // 512x12x12x64 f16
  half_t* C2H  = (half_t*)(ws + 29392896);   // 512x12x12x32 f16
  float*  R24  = (float*)(ws + 34111488);    // 512x24x24x32 f32

  float* y_out = (float*)d_out;              // (512,576)
  float* theta_out = y_out + 512 * 576;      // (512,250)

  // ---- encoder ----
  ln_kernel<<<512, 256, 0, stream>>>(inp, lng, lnb, XN);
  { int total = 512 * 11 * 11 * 32;
    conv_valid_s2_elu_h<24, 24, 1, 11, 11, 32>
        <<<(total + 255) / 256, 256, 0, stream>>>(XN, w1, b1, A1H, total); }
  cvt_f16<<<(512 * 64 + 255) / 256, 256, 0, stream>>>(w2, W2H, 512 * 64);
  convg_s2_elu<11, 11, 32, 4, 4, 64><<<dim3(512, 4), 32, 0, stream>>>(A1H, W2H, b2, A2H);
  cvt_f16<<<(1024 * 128 + 255) / 256, 256, 0, stream>>>(w3, W3H, 1024 * 128);
  convg_s2_elu<4, 4, 64, 1, 1, 128><<<dim3(32, 8), 32, 0, stream>>>(A2H, W3H, b3, A3H);
  cvt_f16<<<(128 * 512 + 255) / 256, 256, 0, stream>>>(we, WEH, 128 * 512);
  gemm16<2, 512, 128, float><<<dim3(32, 32), 32, 0, stream>>>(A3H, WEH, be, X512);   // ELU

  // ---- RNN (single launch for all 512 steps) ----
  pad_f16<<<(256 * 256 + 255) / 256, 256, 0, stream>>>(wrh, WRHH, 250, 250, 256, 256 * 256);
  rnn_kernel<<<32, 512, 0, stream>>>(X512, WRHH, wrin, brv, th0, H16, theta_out);

  // ---- decoder dense ----
  pad_f16<<<(256 * 512 + 255) / 256, 256, 0, stream>>>(wd1, WD1H, 250, 512, 512, 256 * 512);
  gemm16<1, 512, 256, half_t><<<dim3(32, 32), 32, 0, stream>>>(H16, WD1H, bd1, D1H); // ReLU
  cvt_f16<<<(512 * 1152 + 255) / 256, 256, 0, stream>>>(wd2, WD2H, 512 * 1152);
  gemm16<1, 1152, 512, float><<<dim3(32, 72), 32, 0, stream>>>(D1H, WD2H, bd2, D2);  // ReLU

  // ---- decoder convs (implicit-GEMM WMMA) ----
  { int total = 512 * 6 * 6 * 128;
    bilinear2x<float, half_t, 3, 128><<<(total + 255) / 256, 256, 0, stream>>>(D2, R6H, total); }
  cvt_f16<<<(2048 * 64 + 255) / 256, 256, 0, stream>>>(wdc1, WC1H, 2048 * 64);
  convg_same_relu<6, 6, 128, 64><<<dim3(1152, 4), 32, 0, stream>>>(R6H, WC1H, bdc1, C1H);
  { int total = 512 * 12 * 12 * 64;
    bilinear2x<half_t, half_t, 6, 64><<<(total + 255) / 256, 256, 0, stream>>>(C1H, R12H, total); }
  cvt_f16<<<(1024 * 32 + 255) / 256, 256, 0, stream>>>(wdc2, WC2H, 1024 * 32);
  convg_same_relu<12, 12, 64, 32><<<dim3(4608, 2), 32, 0, stream>>>(R12H, WC2H, bdc2, C2H);
  { int total = 512 * 24 * 24 * 32;
    bilinear2x<half_t, float, 12, 32><<<(total + 255) / 256, 256, 0, stream>>>(C2H, R24, total); }
  { int total = 512 * 24 * 24;
    conv_same_s1_f<24, 24, 32, 1><<<(total + 255) / 256, 256, 0, stream>>>(R24, wdc3, bdc3, y_out, total); }
}
